// AdaptiveSpikingAttention_64682207478155
// MI455X (gfx1250) — matrix operand order repaired and checked by reading, not compiled, not run on hardware
//
#include <hip/hip_runtime.h>
#include <hip/hip_bf16.h>

typedef __attribute__((ext_vector_type(16))) _Float16 v16h;
typedef __attribute__((ext_vector_type(2)))  _Float16 v2h;
typedef __attribute__((ext_vector_type(8)))  float    v8f;
typedef __attribute__((ext_vector_type(2)))  float    v2f;

#define B_SZ    4
#define S_LEN   1024
#define E_DIM   512
#define H_NUM   8
#define D_HEAD  64
#define M_ROWS  4096
#define T_MAXC  20

__device__ __forceinline__ v8f wmma_f32x4(v2f a, v2f b, v8f c) {
  return __builtin_amdgcn_wmma_f32_16x16x4_f32(false, a, false, b, (short)0, c, false, false);
}
__device__ __forceinline__ v8f wmma_f16x32(v16h a, v16h b, v8f c) {
  return __builtin_amdgcn_wmma_f32_16x16x32_f16(false, a, false, b, (short)0, c, false, false);
}

// Async global->LDS copy of 16 bytes per lane (ASYNCcnt-tracked).
// lds_off is the LDS byte offset (low 32 bits of a generic pointer to LDS).
__device__ __forceinline__ void async_copy_b128(unsigned lds_off, const void* gsrc) {
  asm volatile("global_load_async_to_lds_b128 %0, %1, off"
               :: "v"(lds_off), "v"(gsrc) : "memory");
}

// ---------------------------------------------------------------------------
// Kernel 1: gate/comp MLPs -> adaptive window T_i per token (block per token)
// ---------------------------------------------------------------------------
__global__ __launch_bounds__(128) void gate_window_kernel(
    const float* __restrict__ x,
    const float* __restrict__ gW1, const float* __restrict__ gb1,
    const float* __restrict__ gg,  const float* __restrict__ gbe,
    const float* __restrict__ gW2, const float* __restrict__ gb2,
    const float* __restrict__ gW3, const float* __restrict__ gb3,
    const float* __restrict__ cW1, const float* __restrict__ cb1,
    const float* __restrict__ cg,  const float* __restrict__ cbe,
    const float* __restrict__ cW2, const float* __restrict__ cb2,
    const float* __restrict__ cW3, const float* __restrict__ cb3,
    int* __restrict__ Ti)
{
  __shared__ float sx[E_DIM];
  __shared__ float sh[128];
  __shared__ float sred[128];
  __shared__ float s_gate;
  const int tid = threadIdx.x;
  const int token = blockIdx.x;

  for (int e = tid; e < E_DIM; e += 128) sx[e] = x[(size_t)token * E_DIM + e];
  __syncthreads();

#define REDUCE_TO(dst, val) do { sred[tid] = (val); __syncthreads();                          \
    for (int off = 64; off > 0; off >>= 1) {                                                  \
      if (tid < off) sred[tid] += sred[tid + off];                                            \
      __syncthreads(); }                                                                      \
    (dst) = sred[0]; __syncthreads(); } while (0)

  // ---- gate branch: 512 -> 128 (LN, ReLU) -> 64 (ReLU) -> 1 (sigmoid)
  float h1 = gb1[tid];
  for (int e = 0; e < E_DIM; ++e) h1 = fmaf(sx[e], gW1[e * 128 + tid], h1);
  float tot;
  REDUCE_TO(tot, h1);
  float mu = tot * (1.0f / 128.0f);
  float dcen = h1 - mu;
  REDUCE_TO(tot, dcen * dcen);
  float var = tot * (1.0f / 128.0f);
  float hn = fmaxf(dcen * rsqrtf(var + 1e-5f) * gg[tid] + gbe[tid], 0.0f);
  sh[tid] = hn;
  __syncthreads();
  float h2 = 0.0f;
  if (tid < 64) {
    h2 = gb2[tid];
    for (int e = 0; e < 128; ++e) h2 = fmaf(sh[e], gW2[e * 64 + tid], h2);
    h2 = fmaxf(h2, 0.0f);
  }
  REDUCE_TO(tot, (tid < 64) ? h2 * gW3[tid] : 0.0f);
  if (tid == 0) s_gate = 1.0f / (1.0f + expf(-(tot + gb3[0])));
  __syncthreads();

  // ---- comp branch: 512 -> 64 (LN, ReLU) -> 32 (ReLU) -> 1 (sigmoid)
  float c1 = 0.0f;
  if (tid < 64) {
    c1 = cb1[tid];
    for (int e = 0; e < E_DIM; ++e) c1 = fmaf(sx[e], cW1[e * 64 + tid], c1);
  }
  REDUCE_TO(tot, (tid < 64) ? c1 : 0.0f);
  float muc = tot * (1.0f / 64.0f);
  float dc = c1 - muc;
  REDUCE_TO(tot, (tid < 64) ? dc * dc : 0.0f);
  float varc = tot * (1.0f / 64.0f);
  if (tid < 64) sh[tid] = fmaxf(dc * rsqrtf(varc + 1e-5f) * cg[tid] + cbe[tid], 0.0f);
  __syncthreads();
  float c2 = 0.0f;
  if (tid < 32) {
    c2 = cb2[tid];
    for (int e = 0; e < 64; ++e) c2 = fmaf(sh[e], cW2[e * 32 + tid], c2);
    c2 = fmaxf(c2, 0.0f);
  }
  REDUCE_TO(tot, (tid < 32) ? c2 * cW3[tid] : 0.0f);
  if (tid == 0) {
    float comp = 1.0f / (1.0f + expf(-(tot + cb3[0])));
    int s = token & (S_LEN - 1);
    float pos = 0.8f + 0.4f * (float)s * (1.0f / 1023.0f);
    float score = (0.7f * s_gate + 0.3f * comp) * pos;
    float tI = ceilf(score * (float)T_MAXC);
    tI = fminf(fmaxf(tI, 1.0f), (float)T_MAXC);
    Ti[token] = (int)tI;
  }
#undef REDUCE_TO
}

// ---------------------------------------------------------------------------
// Kernel 2: fused  X@W  (fp32 WMMA 16x16x4)  +  LIF spike accumulation.
// The 64-wide W panel is staged into LDS once per block via async
// global->LDS DMA (double buffered, 32 k-rows per chunk), so all 8 waves
// share one copy. grid.z: 0->q, 1->k (row-major f16), 2->v (transposed f16).
// Each wave owns a 16(M) x 64(N) C tile.
// ---------------------------------------------------------------------------
__global__ __launch_bounds__(256) void qkv_spike_kernel(
    const float* __restrict__ X,
    const float* __restrict__ Wq, const float* __restrict__ Wk, const float* __restrict__ Wv,
    const float* aq, const float* bq, const float* ak, const float* bk,
    const float* av, const float* bv,
    const int* __restrict__ Ti,
    _Float16* __restrict__ qsum, _Float16* __restrict__ ksum, _Float16* __restrict__ vT)
{
  // 32 k-rows x 64 n-cols, row stride padded to 80 floats (rows 2 apart land
  // 32 banks apart -> conflict-free half-wave B-operand reads), x2 buffers.
  __shared__ float wtile[2][32][80];

  const int z = blockIdx.z;
  const float* W = (z == 0) ? Wq : (z == 1) ? Wk : Wv;
  const float alpha = (z == 0) ? aq[0] : (z == 1) ? ak[0] : av[0];
  const float beta  = (z == 0) ? bq[0] : (z == 1) ? bk[0] : bv[0];

  const int tid  = threadIdx.x;
  const int wave = tid >> 5;
  const int lane = tid & 31;
  const int lm = lane & 15;
  const int hl = lane >> 4;
  const int m0 = blockIdx.x * 128 + wave * 16;
  const int n0 = blockIdx.y * 64;

  const unsigned lds0 = (unsigned)(uintptr_t)&wtile[0][0][0];
  const unsigned lds1 = (unsigned)(uintptr_t)&wtile[1][0][0];

  // Issue async copy of one 32x64 W chunk into buffer `buf`.
  // 512 x 16B segments; each of 256 threads copies 2 segments.
  auto issue_chunk = [&](int buf, int k0) {
#pragma unroll
    for (int ss = 0; ss < 2; ++ss) {
      const int sg   = tid + ss * 256;
      const int row  = sg >> 4;        // 0..31
      const int cseg = sg & 15;        // 16B segment within the 64-float row
      const unsigned lo = (buf ? lds1 : lds0) + (unsigned)(row * 320 + cseg * 16);
      const float* gp = W + (size_t)(k0 + row) * E_DIM + n0 + cseg * 4;
      async_copy_b128(lo, gp);
    }
  };

  const float* arow = X + (size_t)(m0 + lm) * E_DIM;
  v8f cacc[4] = {v8f{}, v8f{}, v8f{}, v8f{}};

  issue_chunk(0, 0);
  for (int kc = 0; kc < E_DIM / 32; ++kc) {
    if (kc + 1 < E_DIM / 32) {
      issue_chunk((kc + 1) & 1, (kc + 1) * 32);
      asm volatile("s_wait_asynccnt 0x2" ::: "memory");  // current chunk done
    } else {
      asm volatile("s_wait_asynccnt 0x0" ::: "memory");
    }
    __syncthreads();

    const float (*wt)[80] = wtile[kc & 1];
#pragma unroll
    for (int kk = 0; kk < 32; kk += 4) {
      const int kglob = kc * 32 + kk;
      v2f a;
      a[0] = arow[kglob + 2 * hl];
      a[1] = arow[kglob + 2 * hl + 1];
      const float* wl = &wt[kk + 2 * hl][lm];
#pragma unroll
      for (int nt = 0; nt < 4; ++nt) {
        v2f b;
        b[0] = wl[nt * 16];        // row k+2*hl
        b[1] = wl[80 + nt * 16];   // row k+2*hl+1
        cacc[nt] = wmma_f32x4(a, b, cacc[nt]);
      }
    }
    __syncthreads();  // all waves done reading before buffer is overwritten
  }

  int trows[8];
#pragma unroll
  for (int i = 0; i < 8; ++i) trows[i] = Ti[m0 + i + 8 * hl];

#pragma unroll
  for (int nt = 0; nt < 4; ++nt) {
    const int col = n0 + nt * 16 + lm;
#pragma unroll
    for (int i = 0; i < 8; ++i) {
      const float xv = cacc[nt][i];
      float ii = 0.0f, vv = 0.0f, acc = 0.0f;
      const int T = trows[i];
      for (int t = 0; t < T; ++t) {          // for t >= T_i, acc is unchanged
        ii = fmaf(alpha, ii, xv);
        vv = fmaf(beta, vv, ii);
        if (vv >= 1.0f) { acc += 1.0f; vv = 0.0f; }
      }
      const int row = m0 + i + 8 * hl;
      if (z == 2)      vT[(size_t)col * M_ROWS + row] = (_Float16)acc;
      else if (z == 0) qsum[(size_t)row * E_DIM + col] = (_Float16)acc;
      else             ksum[(size_t)row * E_DIM + col] = (_Float16)acc;
    }
  }
}

// ---------------------------------------------------------------------------
// Kernel 3: flash attention over spike sums (f16 WMMA, exact integer scores).
// 1 wave = 16 queries x one head; loops 32 keys/iter with online softmax.
// O is accumulated transposed (O^T = V^T @ P^T) so rescale factors are
// uniform per lane. P bounces through LDS to switch C-layout -> B-layout.
// ---------------------------------------------------------------------------
__global__ __launch_bounds__(128) void attn_kernel(
    const _Float16* __restrict__ qsum, const _Float16* __restrict__ ksum,
    const _Float16* __restrict__ vT, float* __restrict__ attn)
{
  __shared__ _Float16 s_p[4][16 * 32];
  __shared__ float    s_f[4][16];

  const int wave = threadIdx.x >> 5;
  const int lane = threadIdx.x & 31;
  const int lm = lane & 15;
  const int hl = lane >> 4;
  const int bh = blockIdx.y;             // b*H + h
  const int b = bh >> 3;
  const int h = bh & 7;
  const int mt = blockIdx.x * 4 + wave;  // 16-query tile index within S
  const int qrow0 = b * S_LEN + mt * 16;

  _Float16* pl = &s_p[wave][0];
  float*    fl = &s_f[wave][0];

  // Load Q tile (16 x 64) as two f16 A operands (D 0..31 and 32..63)
  v16h aq0, aq1;
  {
    const _Float16* qb = qsum + (size_t)(qrow0 + lm) * E_DIM + h * D_HEAD;
#pragma unroll
    for (int v = 0; v < 8; ++v) {
      const int K = (v < 4) ? (2 * v + 8 * hl) : (16 + 2 * (v - 4) + 8 * hl);
      v2h p0 = *(const v2h*)(qb + K);
      v2h p1 = *(const v2h*)(qb + 32 + K);
      aq0[2 * v] = p0[0]; aq0[2 * v + 1] = p0[1];
      aq1[2 * v] = p1[0]; aq1[2 * v + 1] = p1[1];
    }
  }

  v8f o[4] = {v8f{}, v8f{}, v8f{}, v8f{}};
  float mrow[8], lrow[8];
#pragma unroll
  for (int i = 0; i < 8; ++i) { mrow[i] = -1e30f; lrow[i] = 0.0f; }
  const float scale = 0.125f;  // D^-0.5

  for (int j = 0; j < S_LEN / 32; ++j) {
    const int krow0 = b * S_LEN + j * 32;

    // ---- scores: two 16x16 tiles (keys j*32..+15, +16..+31)
    v8f s0 = {}, s1 = {};
    {
      const _Float16* kb0 = ksum + (size_t)(krow0 + lm) * E_DIM + h * D_HEAD;
      const _Float16* kb1 = ksum + (size_t)(krow0 + 16 + lm) * E_DIM + h * D_HEAD;
      v16h b0a, b0b, b1a, b1b;
#pragma unroll
      for (int v = 0; v < 8; ++v) {
        const int K = (v < 4) ? (2 * v + 8 * hl) : (16 + 2 * (v - 4) + 8 * hl);
        v2h p;
        p = *(const v2h*)(kb0 + K);      b0a[2 * v] = p[0]; b0a[2 * v + 1] = p[1];
        p = *(const v2h*)(kb0 + 32 + K); b0b[2 * v] = p[0]; b0b[2 * v + 1] = p[1];
        p = *(const v2h*)(kb1 + K);      b1a[2 * v] = p[0]; b1a[2 * v + 1] = p[1];
        p = *(const v2h*)(kb1 + 32 + K); b1b[2 * v] = p[0]; b1b[2 * v + 1] = p[1];
      }
      s0 = wmma_f16x32(aq0, b0a, s0);
      s0 = wmma_f16x32(aq1, b0b, s0);
      s1 = wmma_f16x32(aq0, b1a, s1);
      s1 = wmma_f16x32(aq1, b1b, s1);
    }

    // ---- online softmax bookkeeping (rows live in score C-layout)
    float mnew[8], corr[8];
#pragma unroll
    for (int i = 0; i < 8; ++i) {
      float a = fmaxf(s0[i], s1[i]) * scale;
      a = fmaxf(a, __shfl_xor(a, 1));
      a = fmaxf(a, __shfl_xor(a, 2));
      a = fmaxf(a, __shfl_xor(a, 4));
      a = fmaxf(a, __shfl_xor(a, 8));
      mnew[i] = fmaxf(mrow[i], a);
      corr[i] = __expf(mrow[i] - mnew[i]);
    }
#pragma unroll
    for (int i = 0; i < 8; ++i) {
      const float p0 = __expf(s0[i] * scale - mnew[i]);
      const float p1 = __expf(s1[i] * scale - mnew[i]);
      const int q = i + 8 * hl;
      pl[q * 32 + lm]      = (_Float16)p0;   // P stored [q][key]
      pl[q * 32 + 16 + lm] = (_Float16)p1;
      float rs = p0 + p1;
      rs += __shfl_xor(rs, 1);
      rs += __shfl_xor(rs, 2);
      rs += __shfl_xor(rs, 4);
      rs += __shfl_xor(rs, 8);
      lrow[i] = lrow[i] * corr[i] + rs;
      mrow[i] = mnew[i];
    }
    if (lm == 0) {
#pragma unroll
      for (int i = 0; i < 8; ++i) fl[8 * hl + i] = corr[i];
    }
    asm volatile("s_wait_dscnt 0" ::: "memory");
    const float f = fl[lm];                 // per-query rescale, uniform per lane
#pragma unroll
    for (int cth = 0; cth < 4; ++cth) {
#pragma unroll
      for (int i = 0; i < 8; ++i) o[cth][i] *= f;
    }

    // ---- P^T as B operand from LDS (pairs of keys are contiguous)
    v16h bp;
#pragma unroll
    for (int v = 0; v < 8; ++v) {
      const int K = (v < 4) ? (2 * v + 8 * hl) : (16 + 2 * (v - 4) + 8 * hl);
      v2h p = *(const v2h*)(pl + lm * 32 + K);
      bp[2 * v] = p[0]; bp[2 * v + 1] = p[1];
    }

    // ---- O^T += V^T @ P^T  (A from transposed v_sum: dword loads)
    const size_t vbase = (size_t)(h * D_HEAD + lm) * M_ROWS + (size_t)b * S_LEN + j * 32;
#pragma unroll
    for (int cth = 0; cth < 4; ++cth) {
      const _Float16* vb = vT + vbase + (size_t)cth * 16 * M_ROWS;
      v16h avv;
#pragma unroll
      for (int v = 0; v < 8; ++v) {
        const int K = (v < 4) ? (2 * v + 8 * hl) : (16 + 2 * (v - 4) + 8 * hl);
        v2h p = *(const v2h*)(vb + K);
        avv[2 * v] = p[0]; avv[2 * v + 1] = p[1];
      }
      o[cth] = wmma_f16x32(avv, bp, o[cth]);
    }
  }

  // ---- finalize: divide by l (softmax denom) and T_MAX (v_mean), store
  if (lm == 0) {
#pragma unroll
    for (int i = 0; i < 8; ++i) fl[8 * hl + i] = lrow[i];
  }
  asm volatile("s_wait_dscnt 0" ::: "memory");
  const float linv = 1.0f / (fl[lm] * (float)T_MAXC);
  float* ab = attn + (size_t)(qrow0 + lm) * E_DIM + h * D_HEAD;
#pragma unroll
  for (int cth = 0; cth < 4; ++cth) {
#pragma unroll
    for (int i = 0; i < 8; ++i) {
      const int d = cth * 16 + i + 8 * hl;
      ab[d] = o[cth][i] * linv;
    }
  }
}

// ---------------------------------------------------------------------------
// Kernel 4: out = attn @ Wo + bo (fp32 WMMA 16x16x4), writes d_out
// ---------------------------------------------------------------------------
__global__ __launch_bounds__(256) void out_gemm_kernel(
    const float* __restrict__ A, const float* __restrict__ Wo,
    const float* __restrict__ bo, float* __restrict__ out)
{
  const int wave = threadIdx.x >> 5;
  const int lane = threadIdx.x & 31;
  const int lm = lane & 15;
  const int hl = lane >> 4;
  const int m0 = blockIdx.x * 128 + wave * 16;
  const int n0 = blockIdx.y * 64;

  const float* arow = A + (size_t)(m0 + lm) * E_DIM;
  v8f cacc[4] = {v8f{}, v8f{}, v8f{}, v8f{}};

  for (int k = 0; k < E_DIM; k += 4) {
    v2f a;
    a[0] = arow[k + 2 * hl];
    a[1] = arow[k + 2 * hl + 1];
    const float* w0 = Wo + (size_t)(k + 2 * hl) * E_DIM + n0 + lm;
    const float* w1 = w0 + E_DIM;
    __builtin_prefetch(w0 + 8 * E_DIM, 0, 0);
#pragma unroll
    for (int nt = 0; nt < 4; ++nt) {
      v2f b;
      b[0] = w0[nt * 16];
      b[1] = w1[nt * 16];
      cacc[nt] = wmma_f32x4(a, b, cacc[nt]);
    }
  }
#pragma unroll
  for (int nt = 0; nt < 4; ++nt) {
    const int col = n0 + nt * 16 + lm;
    const float bias = bo[col];
#pragma unroll
    for (int i = 0; i < 8; ++i) {
      out[(size_t)(m0 + i + 8 * hl) * E_DIM + col] = cacc[nt][i] + bias;
    }
  }
}

// ---------------------------------------------------------------------------
extern "C" void kernel_launch(void* const* d_in, const int* in_sizes, int n_in,
                              void* d_out, int out_size, void* d_ws, size_t ws_size,
                              hipStream_t stream) {
  const float* x   = (const float*)d_in[0];
  const float* Wq  = (const float*)d_in[1];
  const float* Wk  = (const float*)d_in[2];
  const float* Wv  = (const float*)d_in[3];
  const float* Wo  = (const float*)d_in[4];
  const float* bo  = (const float*)d_in[5];
  const float* gW1 = (const float*)d_in[6];
  const float* gb1 = (const float*)d_in[7];
  const float* gg  = (const float*)d_in[8];
  const float* gbe = (const float*)d_in[9];
  const float* gW2 = (const float*)d_in[10];
  const float* gb2 = (const float*)d_in[11];
  const float* gW3 = (const float*)d_in[12];
  const float* gb3 = (const float*)d_in[13];
  const float* cW1 = (const float*)d_in[14];
  const float* cb1 = (const float*)d_in[15];
  const float* cg  = (const float*)d_in[16];
  const float* cbe = (const float*)d_in[17];
  const float* cW2 = (const float*)d_in[18];
  const float* cb2 = (const float*)d_in[19];
  const float* cW3 = (const float*)d_in[20];
  const float* cb3 = (const float*)d_in[21];
  const float* alpha_q = (const float*)d_in[22];
  const float* beta_q  = (const float*)d_in[23];
  const float* alpha_k = (const float*)d_in[24];
  const float* beta_k  = (const float*)d_in[25];
  const float* alpha_v = (const float*)d_in[26];
  const float* beta_v  = (const float*)d_in[27];

  char* ws = (char*)d_ws;
  int*      Ti   = (int*)ws;                                    // 16 KB (reserve 64 KB)
  _Float16* qsum = (_Float16*)(ws + (size_t)(64 << 10));        // 4 MB
  _Float16* ksum = qsum + (size_t)M_ROWS * E_DIM;               // 4 MB
  _Float16* vT   = ksum + (size_t)M_ROWS * E_DIM;               // 4 MB (transposed [E][M])
  float*    attn = (float*)(ws + (size_t)(64 << 10)
                            + 3 * (size_t)M_ROWS * E_DIM * sizeof(_Float16)); // 8 MB

  gate_window_kernel<<<M_ROWS, 128, 0, stream>>>(
      x, gW1, gb1, gg, gbe, gW2, gb2, gW3, gb3,
      cW1, cb1, cg, cbe, cW2, cb2, cW3, cb3, Ti);

  qkv_spike_kernel<<<dim3(M_ROWS / 128, E_DIM / 64, 3), 256, 0, stream>>>(
      x, Wq, Wk, Wv, alpha_q, beta_q, alpha_k, beta_k, alpha_v, beta_v,
      Ti, qsum, ksum, vT);

  attn_kernel<<<dim3((S_LEN / 16) / 4, B_SZ * H_NUM), 128, 0, stream>>>(
      qsum, ksum, vT, attn);

  out_gemm_kernel<<<dim3(M_ROWS / 128, E_DIM / 64), 256, 0, stream>>>(
      attn, Wo, bo, (float*)d_out);
}
